// MSCAT_83940840833462
// MI455X (gfx1250) — matrix operand order repaired and checked by reading, not compiled, not run on hardware
//
#include <hip/hip_runtime.h>
#include <hip/hip_bf16.h>
#include <math.h>

// ---- problem constants (match the reference) ----
#define Bc    4
#define Sc    4096
#define Dc    512
#define Hc    8
#define Wc    32
#define HDc   64
#define Mfull (Bc * Sc)     // 16384
#define KSZ   7

typedef __bf16 bf16_t;
typedef __attribute__((ext_vector_type(16))) __bf16 v16bf_t;
typedef __attribute__((ext_vector_type(8)))  __bf16 v8bf_t;
typedef __attribute__((ext_vector_type(8)))  float  v8f_t;

union AFrag { v16bf_t v; v8bf_t h[2]; };

__device__ __forceinline__ float gelu_tanh(float x) {
  const float c0 = 0.7978845608028654f;  // sqrt(2/pi)
  float x3 = x * x * x;
  return 0.5f * x * (1.0f + tanhf(c0 * (x + 0.044715f * x3)));
}

// =====================================================================
// Tiled GEMM: Y[M,N] = act(X[M,K] @ W[K,N] + bias), fp32 in/out,
// bf16 WMMA compute. Block tile 128x128, BK=32, 256 threads = 8 waves
// in a 4(M) x 2(N) grid; each wave owns a 32x64 tile = 2x4 WMMA frags.
// Double-buffered LDS; next K-tile staged in registers while WMMAs run.
// =====================================================================
#define BM 128
#define BN 128
#define BK 32
#define APAD 40   // bf16 elems per A row in LDS (32 + pad)
#define BPAD 40   // bf16 elems per Bt row in LDS

__device__ __forceinline__ void stage_load(
    const float* __restrict__ X, const float* __restrict__ Wm,
    int tileM, int tileN, int K, int N, int k0,
    int xr, int xc, int wk, int wc4, float4 (&xa)[4], float4 (&wa)[4])
{
#pragma unroll
  for (int p = 0; p < 4; ++p)
    xa[p] = *(const float4*)(X + (size_t)(tileM + xr + p * 32) * K + k0 + xc * 4);
#pragma unroll
  for (int p = 0; p < 4; ++p)
    wa[p] = *(const float4*)(Wm + (size_t)(k0 + wk + p * 8) * N + tileN + wc4 * 4);
}

__device__ __forceinline__ void stage_store(
    bf16_t* __restrict__ As, bf16_t* __restrict__ Bs,
    int xr, int xc, int wk, int wc4, const float4 (&xa)[4], const float4 (&wa)[4])
{
#pragma unroll
  for (int p = 0; p < 4; ++p) {
    union { bf16_t e[4]; uint2 u; } pk;
    pk.e[0] = (bf16_t)xa[p].x; pk.e[1] = (bf16_t)xa[p].y;
    pk.e[2] = (bf16_t)xa[p].z; pk.e[3] = (bf16_t)xa[p].w;
    *(uint2*)(&As[(xr + p * 32) * APAD + xc * 4]) = pk.u;
  }
#pragma unroll
  for (int p = 0; p < 4; ++p) {
    const int kk = wk + p * 8;          // transposed store: Bs[n][k]
    Bs[(wc4 * 4 + 0) * BPAD + kk] = (bf16_t)wa[p].x;
    Bs[(wc4 * 4 + 1) * BPAD + kk] = (bf16_t)wa[p].y;
    Bs[(wc4 * 4 + 2) * BPAD + kk] = (bf16_t)wa[p].z;
    Bs[(wc4 * 4 + 3) * BPAD + kk] = (bf16_t)wa[p].w;
  }
}

__global__ __launch_bounds__(256)
void tfm_gemm(const float* __restrict__ X, const float* __restrict__ Wm,
              const float* __restrict__ bias, float* __restrict__ Y,
              int M, int N, int K, int act)
{
  __shared__ __align__(16) bf16_t As[2 * BM * APAD];
  __shared__ __align__(16) bf16_t Bs[2 * BN * BPAD];

  const int tid   = threadIdx.x;
  const int tileM = blockIdx.y * BM;
  const int tileN = blockIdx.x * BN;
  const int wave  = tid >> 5;
  const int lane  = tid & 31;
  const int wm    = wave >> 1;   // 0..3 -> M sub-tile (32 rows)
  const int wn    = wave & 1;    // 0..1 -> N sub-tile (64 cols)
  const int lr    = lane & 15;
  const int lhi   = lane >> 4;

  // staging thread mapping
  const int xr  = tid >> 3;      // 0..31  A row within pass
  const int xc  = tid & 7;       // 0..7   A float4 column
  const int wk  = tid >> 5;      // 0..7   W k-row within pass
  const int wc4 = tid & 31;      // 0..31  W float4 column along N

  v8f_t acc[2][4];
#pragma unroll
  for (int i = 0; i < 2; ++i)
#pragma unroll
    for (int j = 0; j < 4; ++j)
#pragma unroll
      for (int e = 0; e < 8; ++e) acc[i][j][e] = 0.f;

  float4 xa[4], wa[4];
  stage_load(X, Wm, tileM, tileN, K, N, 0, xr, xc, wk, wc4, xa, wa);
  stage_store(As, Bs, xr, xc, wk, wc4, xa, wa);
  __syncthreads();

  int cur = 0;
  for (int k0 = 0; k0 < K; k0 += BK) {
    const bool more = (k0 + BK) < K;
    if (more)  // issue next tile's global loads; they fly during the WMMAs
      stage_load(X, Wm, tileM, tileN, K, N, k0 + BK, xr, xc, wk, wc4, xa, wa);

    // fragments per ISA 16-bit layouts:
    // A: lane(lr,lhi) row=lr, K chunks [lhi*8..+7] and [16+lhi*8..+7]
    // B: lane(lr,lhi) col=lr, K chunks [lhi*8..+7] and [16+lhi*8..+7]
    const bf16_t* Acur = As + cur * (BM * APAD);
    const bf16_t* Bcur = Bs + cur * (BN * BPAD);
    AFrag af[2], bfr[4];
#pragma unroll
    for (int i = 0; i < 2; ++i) {
      const bf16_t* pa = Acur + (wm * 32 + i * 16 + lr) * APAD;
      af[i].h[0] = *(const v8bf_t*)(pa + lhi * 8);
      af[i].h[1] = *(const v8bf_t*)(pa + 16 + lhi * 8);
    }
#pragma unroll
    for (int j = 0; j < 4; ++j) {
      const bf16_t* pb = Bcur + (wn * 64 + j * 16 + lr) * BPAD;
      bfr[j].h[0] = *(const v8bf_t*)(pb + lhi * 8);
      bfr[j].h[1] = *(const v8bf_t*)(pb + 16 + lhi * 8);
    }
#pragma unroll
    for (int i = 0; i < 2; ++i)
#pragma unroll
      for (int j = 0; j < 4; ++j)
        acc[i][j] = __builtin_amdgcn_wmma_f32_16x16x32_bf16(
            false, af[i].v, false, bfr[j].v, (short)0, acc[i][j], false, false);

    if (more) {
      stage_store(As + (1 - cur) * (BM * APAD), Bs + (1 - cur) * (BN * BPAD),
                  xr, xc, wk, wc4, xa, wa);
      __syncthreads();
      cur ^= 1;
    }
  }

  // epilogue: C layout -> lane col = lr, VGPR e holds row e + lhi*8
#pragma unroll
  for (int i = 0; i < 2; ++i)
#pragma unroll
    for (int j = 0; j < 4; ++j) {
      const int gm0 = tileM + wm * 32 + i * 16 + lhi * 8;
      const int gn  = tileN + wn * 64 + j * 16 + lr;
      const float bv = bias ? bias[gn] : 0.f;
#pragma unroll
      for (int e = 0; e < 8; ++e) {
        float val = acc[i][j][e] + bv;
        if (act) val = gelu_tanh(val);
        Y[(size_t)(gm0 + e) * N + gn] = val;
      }
    }
}

// =====================================================================
// Local windowed attention: W=32 window, relative-position bias.
// Block = 1 window (256 thr = 8 waves = 8 heads); lane = query row.
// Two-pass softmax (max pass, then exp pass) with score recomputation (all state in registers).
// =====================================================================
__global__ __launch_bounds__(256)
void tfm_local_attn(const float* __restrict__ qkv, const float* __restrict__ rel,
                    float* __restrict__ o)
{
  const int win = blockIdx.x;               // 0 .. B*(S/W)-1
  const int h = threadIdx.x >> 5;
  const int r = threadIdx.x & 31;
  const int b = win >> 7;                   // S/W = 128 windows per batch
  const int s0 = (win & 127) * Wc;
  const float scale = 0.125f;               // HD^-0.5
  const size_t rowB = 3 * Dc;

  const float* qp = qkv + (size_t)(b * Sc + s0 + r) * rowB + h * HDc;
  float q[HDc];
#pragma unroll
  for (int d = 0; d < HDc; ++d) q[d] = qp[d];

  float mx = -3.0e38f;
  for (int j = 0; j < Wc; ++j) {
    const float* kp = qkv + (size_t)(b * Sc + s0 + j) * rowB + Dc + h * HDc;
    float s = 0.f;
#pragma unroll
    for (int d = 0; d < HDc; ++d) s = fmaf(q[d], kp[d], s);
    s = s * scale + rel[(size_t)(r - j + Wc - 1) * Hc + h];
    mx = fmaxf(mx, s);
  }
  float accv[HDc];
#pragma unroll
  for (int d = 0; d < HDc; ++d) accv[d] = 0.f;
  float sum = 0.f;
  for (int j = 0; j < Wc; ++j) {
    const float* kp = qkv + (size_t)(b * Sc + s0 + j) * rowB + Dc + h * HDc;
    float s = 0.f;
#pragma unroll
    for (int d = 0; d < HDc; ++d) s = fmaf(q[d], kp[d], s);
    s = s * scale + rel[(size_t)(r - j + Wc - 1) * Hc + h];
    const float e = __expf(s - mx);
    sum += e;
    const float* vp = kp + Dc;   // v slice
#pragma unroll
    for (int d = 0; d < HDc; ++d) accv[d] = fmaf(e, vp[d], accv[d]);
  }
  const float inv = 1.f / sum;
  float* op = o + (size_t)(b * Sc + s0 + r) * Dc + h * HDc;
#pragma unroll
  for (int d = 0; d < HDc; ++d) op[d] = accv[d] * inv;
}

// =====================================================================
// Global attention on the strided sequence (128 tokens/batch).
// Block per (b,h,query row); 128 threads; softmax via LDS.
// =====================================================================
__global__ __launch_bounds__(128)
void tfm_global_attn(const float* __restrict__ qkv, float* __restrict__ o)
{
  int idx = blockIdx.x;
  const int r = idx & 127; idx >>= 7;
  const int h = idx & 7;
  const int b = idx >> 3;
  const int t = threadIdx.x;

  __shared__ float qrow[HDc];
  __shared__ float red[128];
  __shared__ float probs[128];

  const size_t base = (size_t)b * 128 * (3 * Dc);
  if (t < HDc) qrow[t] = qkv[base + (size_t)r * (3 * Dc) + h * HDc + t];
  __syncthreads();

  const float* kp = qkv + base + (size_t)t * (3 * Dc) + Dc + h * HDc;
  float s = 0.f;
#pragma unroll
  for (int d = 0; d < HDc; ++d) s = fmaf(qrow[d], kp[d], s);
  s *= 0.125f;
  red[t] = s;
  __syncthreads();
  for (int off = 64; off > 0; off >>= 1) {
    if (t < off) red[t] = fmaxf(red[t], red[t + off]);
    __syncthreads();
  }
  const float mx = red[0];
  __syncthreads();
  const float e = __expf(s - mx);
  probs[t] = e;
  red[t] = e;
  __syncthreads();
  for (int off = 64; off > 0; off >>= 1) {
    if (t < off) red[t] += red[t + off];
    __syncthreads();
  }
  const float inv = 1.f / red[0];
  if (t < HDc) {
    float acc = 0.f;
    const float* vbase = qkv + base + 2 * Dc + h * HDc + t;
    for (int j = 0; j < 128; ++j) acc = fmaf(probs[j], vbase[(size_t)j * (3 * Dc)], acc);
    o[((size_t)b * 128 + r) * Dc + h * HDc + t] = acc * inv;
  }
}

// =====================================================================
// Fused (weighted) residual add + LayerNorm. Wave per row of 512.
// If swa!=null: weights = softmax([*swa, *swb]) (final fusion), else 1,1.
// If x1==null: LN(x0) only.
// =====================================================================
__global__ __launch_bounds__(256)
void tfm_ln(const float* __restrict__ x0, const float* __restrict__ x1,
            const float* __restrict__ swa, const float* __restrict__ swb,
            const float* __restrict__ g, const float* __restrict__ bb,
            float* __restrict__ y, int rows)
{
  const int wave = threadIdx.x >> 5;
  const int lane = threadIdx.x & 31;
  const int row = blockIdx.x * 8 + wave;
  if (row >= rows) return;

  float w0 = 1.f, w1 = 1.f;
  if (swa) {
    const float a = swa[0], c = swb[0];
    const float m = fmaxf(a, c);
    const float ea = __expf(a - m), ec = __expf(c - m);
    const float inv = 1.f / (ea + ec);
    w0 = ea * inv; w1 = ec * inv;
  }
  const float* p0 = x0 + (size_t)row * Dc;
  const float* p1 = x1 ? (x1 + (size_t)row * Dc) : nullptr;

  float v[16];
  float s = 0.f;
#pragma unroll
  for (int i = 0; i < 16; ++i) {
    float t = w0 * p0[lane + i * 32];
    if (p1) t += w1 * p1[lane + i * 32];
    v[i] = t; s += t;
  }
#pragma unroll
  for (int off = 16; off > 0; off >>= 1) s += __shfl_xor(s, off, 32);
  const float mu = s * (1.f / (float)Dc);
  float var = 0.f;
#pragma unroll
  for (int i = 0; i < 16; ++i) { const float d = v[i] - mu; var = fmaf(d, d, var); }
#pragma unroll
  for (int off = 16; off > 0; off >>= 1) var += __shfl_xor(var, off, 32);
  var *= (1.f / (float)Dc);
  const float inv = rsqrtf(var + 1e-5f);
#pragma unroll
  for (int i = 0; i < 16; ++i) {
    const int c = lane + i * 32;
    y[(size_t)row * Dc + c] = (v[i] - mu) * inv * g[c] + bb[c];
  }
}

// depthwise conv1d (K=7, zero pad 3) + residual: y = x + conv(x)
__global__ __launch_bounds__(256)
void tfm_dwconv(const float* __restrict__ x, const float* __restrict__ w,
                const float* __restrict__ bias, float* __restrict__ y, int total)
{
  const int idx = blockIdx.x * 256 + threadIdx.x;
  if (idx >= total) return;
  const int d = idx & (Dc - 1);
  const int s = (idx >> 9) & (Sc - 1);
  const int b = idx >> 21;
  float acc = bias[d];
#pragma unroll
  for (int k = 0; k < KSZ; ++k) {
    const int ss = s + k - (KSZ - 1) / 2;
    if (ss >= 0 && ss < Sc)
      acc = fmaf(x[((size_t)b * Sc + ss) * Dc + d], w[d * KSZ + k], acc);
  }
  y[idx] = x[idx] + acc;
}

// xs[b, s2, d] = xg[b, s2*32, d]
__global__ __launch_bounds__(256)
void tfm_gather32(const float* __restrict__ xg, float* __restrict__ xs)
{
  const int idx = blockIdx.x * 256 + threadIdx.x;   // < 4*128*512
  const int d = idx & (Dc - 1);
  const int s2 = (idx >> 9) & 127;
  const int b = idx >> 16;
  xs[idx] = xg[((size_t)b * Sc + s2 * 32) * Dc + d];
}

// y = xg + lin_interp(ps, S)  (ps is (B,128,D))
__global__ __launch_bounds__(256)
void tfm_interp_add(const float* __restrict__ xg, const float* __restrict__ ps,
                    float* __restrict__ y, int total)
{
  const int idx = blockIdx.x * 256 + threadIdx.x;
  if (idx >= total) return;
  const int d = idx & (Dc - 1);
  const int s = (idx >> 9) & (Sc - 1);
  const int b = idx >> 21;
  float pos = ((float)s + 0.5f) * (128.0f / (float)Sc) - 0.5f;
  pos = fminf(fmaxf(pos, 0.f), 127.f);
  const int i0 = (int)floorf(pos);
  const int i1 = min(i0 + 1, 127);
  const float w = pos - (float)i0;
  const float v = ps[((size_t)b * 128 + i0) * Dc + d] * (1.f - w) +
                  ps[((size_t)b * 128 + i1) * Dc + d] * w;
  y[idx] = xg[idx] + v;
}

// =====================================================================
extern "C" void kernel_launch(void* const* d_in, const int* in_sizes, int n_in,
                              void* d_out, int out_size, void* d_ws, size_t ws_size,
                              hipStream_t stream) {
  (void)in_sizes; (void)n_in; (void)out_size; (void)ws_size;
  const float* x       = (const float*)d_in[0];
  const float* lqkv_w  = (const float*)d_in[1];
  const float* lqkv_b  = (const float*)d_in[2];
  const float* lproj_w = (const float*)d_in[3];
  const float* lproj_b = (const float*)d_in[4];
  const float* lrel    = (const float*)d_in[5];
  const float* lconv_w = (const float*)d_in[6];
  const float* lconv_b = (const float*)d_in[7];
  const float* ln1_g   = (const float*)d_in[8];
  const float* ln1_b   = (const float*)d_in[9];
  const float* lffn_w1 = (const float*)d_in[10];
  const float* lffn_b1 = (const float*)d_in[11];
  const float* lffn_w2 = (const float*)d_in[12];
  const float* lffn_b2 = (const float*)d_in[13];
  const float* ln2_g   = (const float*)d_in[14];
  const float* ln2_b   = (const float*)d_in[15];
  const float* gqkv_w  = (const float*)d_in[16];
  const float* gqkv_b  = (const float*)d_in[17];
  const float* gproj_w = (const float*)d_in[18];
  const float* gproj_b = (const float*)d_in[19];
  const float* gn1_g   = (const float*)d_in[20];
  const float* gn1_b   = (const float*)d_in[21];
  const float* gffn_w1 = (const float*)d_in[22];
  const float* gffn_b1 = (const float*)d_in[23];
  const float* gffn_w2 = (const float*)d_in[24];
  const float* gffn_b2 = (const float*)d_in[25];
  const float* gn2_g   = (const float*)d_in[26];
  const float* gn2_b   = (const float*)d_in[27];
  const float* fw_local  = (const float*)d_in[28];
  const float* fw_global = (const float*)d_in[29];
  const float* fn_g    = (const float*)d_in[30];
  const float* fn_b    = (const float*)d_in[31];

  // workspace carve-up (floats)
  const size_t NXG = (size_t)Mfull * Dc;            // 8388608
  float* wsf  = (float*)d_ws;
  float* bufX = wsf;                                 // xl
  float* bufG = bufX + NXG;                          // xg
  float* T1   = bufG + NXG;
  float* T2   = T1 + NXG;
  float* Cbuf = T2 + NXG;                            // qkv (96MB) / ffn hidden (128MB)
  float* xs   = Cbuf + (size_t)Mfull * (4 * Dc);
  float* qkvs = xs + (size_t)Bc * 128 * Dc;
  float* os   = qkvs + (size_t)Bc * 128 * (3 * Dc);
  float* ps   = os + (size_t)Bc * 128 * Dc;

  hipMemcpyAsync(bufX, x, NXG * sizeof(float), hipMemcpyDeviceToDevice, stream);
  hipMemcpyAsync(bufG, x, NXG * sizeof(float), hipMemcpyDeviceToDevice, stream);

  auto gemm = [&](const float* A, const float* Wm, const float* bias, float* Yp,
                  int M, int N, int K, int act) {
    dim3 grid(N / BN, M / BM);
    tfm_gemm<<<grid, 256, 0, stream>>>(A, Wm, bias, Yp, M, N, K, act);
  };
  const int lnBlocks = Mfull / 8;
  const int elBlocks = (Mfull * Dc) / 256;

  // ---------------- local branch (4 layers) ----------------
  for (int i = 0; i < 4; ++i) {
    const float* qw  = lqkv_w  + (size_t)i * Dc * 3 * Dc;
    const float* qb  = lqkv_b  + (size_t)i * 3 * Dc;
    const float* pw  = lproj_w + (size_t)i * Dc * Dc;
    const float* pb  = lproj_b + (size_t)i * Dc;
    const float* rl  = lrel    + (size_t)i * (2 * Wc - 1) * Hc;
    const float* cw  = lconv_w + (size_t)i * Dc * KSZ;
    const float* cb  = lconv_b + (size_t)i * Dc;
    const float* g1  = ln1_g   + (size_t)i * Dc;
    const float* b1  = ln1_b   + (size_t)i * Dc;
    const float* f1w = lffn_w1 + (size_t)i * Dc * 4 * Dc;
    const float* f1b = lffn_b1 + (size_t)i * 4 * Dc;
    const float* f2w = lffn_w2 + (size_t)i * 4 * Dc * Dc;
    const float* f2b = lffn_b2 + (size_t)i * Dc;
    const float* g2  = ln2_g   + (size_t)i * Dc;
    const float* b2  = ln2_b   + (size_t)i * Dc;

    gemm(bufX, qw, qb, Cbuf, Mfull, 3 * Dc, Dc, 0);                       // qkv
    tfm_local_attn<<<Bc * (Sc / Wc), 256, 0, stream>>>(Cbuf, rl, T1);     // attn -> o
    gemm(T1, pw, pb, T2, Mfull, Dc, Dc, 0);                               // proj
    tfm_ln<<<lnBlocks, 256, 0, stream>>>(bufX, T2, nullptr, nullptr, g1, b1, bufX, Mfull);
    tfm_dwconv<<<elBlocks, 256, 0, stream>>>(bufX, cw, cb, T1, Mfull * Dc);
    gemm(T1, f1w, f1b, Cbuf, Mfull, 4 * Dc, Dc, 1);                       // ffn1 + gelu
    gemm(Cbuf, f2w, f2b, T2, Mfull, Dc, 4 * Dc, 0);                       // ffn2
    tfm_ln<<<lnBlocks, 256, 0, stream>>>(T1, T2, nullptr, nullptr, g2, b2, bufX, Mfull);
  }

  // ---------------- global branch (3 layers) ----------------
  for (int i = 0; i < 3; ++i) {
    const float* qw  = gqkv_w  + (size_t)i * Dc * 3 * Dc;
    const float* qb  = gqkv_b  + (size_t)i * 3 * Dc;
    const float* pw  = gproj_w + (size_t)i * Dc * Dc;
    const float* pb  = gproj_b + (size_t)i * Dc;
    const float* g1  = gn1_g   + (size_t)i * Dc;
    const float* b1  = gn1_b   + (size_t)i * Dc;
    const float* f1w = gffn_w1 + (size_t)i * Dc * 4 * Dc;
    const float* f1b = gffn_b1 + (size_t)i * 4 * Dc;
    const float* f2w = gffn_w2 + (size_t)i * 4 * Dc * Dc;
    const float* f2b = gffn_b2 + (size_t)i * Dc;
    const float* g2  = gn2_g   + (size_t)i * Dc;
    const float* b2  = gn2_b   + (size_t)i * Dc;

    tfm_gather32<<<(Bc * 128 * Dc) / 256, 256, 0, stream>>>(bufG, xs);    // stride-32 sample
    gemm(xs, qw, qb, qkvs, Bc * 128, 3 * Dc, Dc, 0);                      // qkv (small)
    tfm_global_attn<<<Bc * Hc * 128, 128, 0, stream>>>(qkvs, os);         // attn
    gemm(os, pw, pb, ps, Bc * 128, Dc, Dc, 0);                            // proj (small)
    tfm_interp_add<<<elBlocks, 256, 0, stream>>>(bufG, ps, T1, Mfull * Dc);
    tfm_ln<<<lnBlocks, 256, 0, stream>>>(T1, nullptr, nullptr, nullptr, g1, b1, bufG, Mfull);
    gemm(bufG, f1w, f1b, Cbuf, Mfull, 4 * Dc, Dc, 1);                     // ffn1 + gelu
    gemm(Cbuf, f2w, f2b, T2, Mfull, Dc, 4 * Dc, 0);                       // ffn2
    tfm_ln<<<lnBlocks, 256, 0, stream>>>(bufG, T2, nullptr, nullptr, g2, b2, bufG, Mfull);
  }

  // ---------------- final fusion ----------------
  tfm_ln<<<lnBlocks, 256, 0, stream>>>(bufX, bufG, fw_local, fw_global,
                                       fn_g, fn_b, (float*)d_out, Mfull);
}
